// RelationalProposalModule_63402307224292
// MI455X (gfx1250) — matrix operand order repaired and checked by reading, not compiled, not run on hardware
//
#include <hip/hip_runtime.h>

// Problem constants (match reference)
#define BB  8
#define KKE 256   // K (proposals)
#define DD  768   // appearance dim
#define DKK 96    // key/value dim per head
#define HH  8     // heads

typedef __attribute__((ext_vector_type(16))) __bf16 v16bf;
typedef __attribute__((ext_vector_type(8)))  __bf16 v8bf;
typedef __attribute__((ext_vector_type(8)))  float  v8f;

// ---------------- WMMA fragment loaders (per CDNA5 ISA 7.12.2) ----------------
// A (16x32 bf16): lanes 0-15 row M=lane, K=k0+0..7 (v0-3) and k0+16..23 (v4-7);
//                 lanes 16-31 row M=lane-16, K=k0+8..15 and k0+24..31.
__device__ __forceinline__ v16bf load_a_frag(const __bf16* A, int lda, int row0, int k0, int lane) {
    int m  = row0 + (lane & 15);
    int kb = k0 + ((lane & 16) ? 8 : 0);
    const __bf16* p = A + (size_t)m * lda + kb;
    v8bf lo = *(const v8bf*)(p);
    v8bf hi = *(const v8bf*)(p + 16);
    v16bf f;
#pragma unroll
    for (int i = 0; i < 8; ++i) { f[i] = lo[i]; f[i + 8] = hi[i]; }
    return f;
}

// B (32x16 bf16), Bt stored row-major [N][Kdim] (i.e. element (k,n) = Bt[n][k]):
// lanes 0-15 col N=lane hold K=k0+0..7 and k0+16..23; lanes 16-31 hold K=k0+8..15 and k0+24..31.
__device__ __forceinline__ v16bf load_b_frag(const __bf16* Bt, int ldb, int col0, int k0, int lane) {
    int n  = col0 + (lane & 15);
    int kb = k0 + ((lane & 16) ? 8 : 0);
    const __bf16* p = Bt + (size_t)n * ldb + kb;
    v8bf lo = *(const v8bf*)(p);
    v8bf hi = *(const v8bf*)(p + 16);
    v16bf f;
#pragma unroll
    for (int i = 0; i < 8; ++i) { f[i] = lo[i]; f[i + 8] = hi[i]; }
    return f;
}

// ---------------- K0a: f32 -> bf16 cast ----------------
__global__ void cast_bf16_kernel(const float* __restrict__ in, __bf16* __restrict__ out, int n) {
    int i = blockIdx.x * blockDim.x + threadIdx.x;
    if (i < n) out[i] = (__bf16)in[i];
}

// ---------------- K0b: weight transpose+cast: [H][D][DK] f32 -> [H][DK][D] bf16 ----------------
__global__ void wtrans_bf16_kernel(const float* __restrict__ w, __bf16* __restrict__ wt) {
    int i = blockIdx.x * blockDim.x + threadIdx.x;
    if (i >= HH * DD * DKK) return;
    int h = i / (DD * DKK);
    int r = i % (DD * DKK);
    int d = r / DKK;
    int e = r % DKK;
    wt[((size_t)h * DKK + e) * DD + d] = (__bf16)w[i];
}

// ---------------- K1: geo bias  logits[b,h,m,n] = log(max(pe.WGw + b, 1e-6)) ----------------
// HBM-bound: reads the 201MB position_embedding exactly once (8 FMA per loaded float).
__global__ void geo_bias_kernel(const float* __restrict__ pe, const float* __restrict__ WGw,
                                const float* __restrict__ WGb, float* __restrict__ logits) {
    __shared__ float sw[HH * 96];
    __shared__ float sb[HH];
    int tid = threadIdx.x;
    for (int i = tid; i < HH * 96; i += blockDim.x) sw[i] = WGw[i];
    if (tid < HH) sb[tid] = WGb[tid];
    __syncthreads();

    int gid = blockIdx.x * blockDim.x + tid;              // over B*K*K
    if (gid >= BB * KKE * KKE) return;
    const float* p = pe + (size_t)gid * 96;
    float acc[HH];
#pragma unroll
    for (int h = 0; h < HH; ++h) acc[h] = 0.f;
    for (int g = 0; g < 96; g += 4) {
        float4 v = *(const float4*)(p + g);
#pragma unroll
        for (int h = 0; h < HH; ++h) {
            acc[h] = fmaf(v.x, sw[h * 96 + g + 0], acc[h]);
            acc[h] = fmaf(v.y, sw[h * 96 + g + 1], acc[h]);
            acc[h] = fmaf(v.z, sw[h * 96 + g + 2], acc[h]);
            acc[h] = fmaf(v.w, sw[h * 96 + g + 3], acc[h]);
        }
    }
    int b  = gid / (KKE * KKE);
    int mn = gid % (KKE * KKE);
#pragma unroll
    for (int h = 0; h < HH; ++h) {
        float w = fmaxf(acc[h] + sb[h], 1e-6f);           // relu + clip(1e-6) in one max
        logits[(size_t)(b * HH + h) * KKE * KKE + mn] = __logf(w);
    }
}

// ---------------- K2: Q/K/V projections via WMMA bf16 ----------------
// k,q stored [b,h,m,e] bf16; v stored transposed [b,h,e,m] bf16 (for final GEMM B-frag).
__global__ void qkv_gemm_kernel(const __bf16* __restrict__ fabf,
                                const __bf16* __restrict__ wkt, const __bf16* __restrict__ wqt,
                                const __bf16* __restrict__ wvt,
                                const float* __restrict__ kb_, const float* __restrict__ qb_,
                                const float* __restrict__ vb_,
                                __bf16* __restrict__ kout, __bf16* __restrict__ qout,
                                __bf16* __restrict__ vtout) {
    int wid  = (blockIdx.x * blockDim.x + threadIdx.x) >> 5;
    int lane = threadIdx.x & 31;
    const int TILES = 16 * 6;                      // (K/16) x (DK/16) per (b,h)
    const int TOTAL = 3 * BB * HH * TILES;
    if (wid >= TOTAL) return;
    int w  = wid / (BB * HH * TILES);
    int r  = wid % (BB * HH * TILES);
    int bh = r / TILES;
    int t  = r % TILES;
    int b  = bh / HH, h = bh % HH;
    int mt = t / 6, et = t % 6;

    const __bf16* A    = fabf + (size_t)b * KKE * DD;
    const __bf16* Bt   = (w == 0 ? wkt : w == 1 ? wqt : wvt) + (size_t)h * DKK * DD;
    const float*  bias = (w == 0 ? kb_ : w == 1 ? qb_ : vb_) + h * DKK;

    v8f acc = {};
    for (int k0 = 0; k0 < DD; k0 += 32) {
        v16bf a  = load_a_frag(A, DD, mt * 16, k0, lane);
        v16bf bf = load_b_frag(Bt, DD, et * 16, k0, lane);
        acc = __builtin_amdgcn_wmma_f32_16x16x32_bf16(false, a, false, bf, (short)0, acc, false, false);
    }
    int n     = et * 16 + (lane & 15);
    int mbase = mt * 16 + ((lane & 16) ? 8 : 0);
    float bv  = bias[n];
    if (w < 2) {
        __bf16* O = (w == 0 ? kout : qout) + (size_t)bh * KKE * DKK;
#pragma unroll
        for (int j = 0; j < 8; ++j) O[(size_t)(mbase + j) * DKK + n] = (__bf16)(acc[j] + bv);
    } else {
        __bf16* O = vtout + (size_t)bh * DKK * KKE;     // [e][m]
#pragma unroll
        for (int j = 0; j < 8; ++j) O[(size_t)n * KKE + mbase + j] = (__bf16)(acc[j] + bv);
    }
}

// ---------------- K3: scores = k.q^T / sqrt(DK), accumulated in-place onto geo bias ----------------
__global__ void scores_kernel(const __bf16* __restrict__ kbf, const __bf16* __restrict__ qbf,
                              float* __restrict__ logits) {
    int wid  = (blockIdx.x * blockDim.x + threadIdx.x) >> 5;
    int lane = threadIdx.x & 31;
    const int TILES = 16 * 16;
    if (wid >= BB * HH * TILES) return;
    int bh = wid / TILES, t = wid % TILES;
    int mt = t / 16, nt = t % 16;
    const __bf16* A  = kbf + (size_t)bh * KKE * DKK;   // rows m, contraction e
    const __bf16* Bt = qbf + (size_t)bh * KKE * DKK;   // element (e,n) = q[n][e]
    v8f acc = {};
#pragma unroll
    for (int k0 = 0; k0 < DKK; k0 += 32) {
        v16bf a = load_a_frag(A, DKK, mt * 16, k0, lane);
        v16bf b = load_b_frag(Bt, DKK, nt * 16, k0, lane);
        acc = __builtin_amdgcn_wmma_f32_16x16x32_bf16(false, a, false, b, (short)0, acc, false, false);
    }
    const float scale = 0.1020620726159658f;           // 1/sqrt(96)
    float* L  = logits + (size_t)bh * KKE * KKE;
    int n     = nt * 16 + (lane & 15);
    int mbase = mt * 16 + ((lane & 16) ? 8 : 0);
#pragma unroll
    for (int j = 0; j < 8; ++j) {
        size_t idx = (size_t)(mbase + j) * KKE + n;
        L[idx] += acc[j] * scale;
    }
}

// ---------------- K4: softmax over m (column-wise) -> attn^T bf16 [b,h,n,m] ----------------
__global__ void softmax_kernel(const float* __restrict__ logits, __bf16* __restrict__ attnT) {
    int bh = blockIdx.x;
    int n  = threadIdx.x;
    const float* L = logits + (size_t)bh * KKE * KKE + n;
    float mx = -3.4e38f;
    for (int m = 0; m < KKE; ++m) mx = fmaxf(mx, L[(size_t)m * KKE]);
    float s = 0.f;
    for (int m = 0; m < KKE; ++m) s += __expf(L[(size_t)m * KKE] - mx);
    float inv = 1.f / s;
    __bf16* AT = attnT + (size_t)bh * KKE * KKE + (size_t)n * KKE;
    for (int m = 0; m < KKE; ++m) AT[m] = (__bf16)(__expf(L[(size_t)m * KKE] - mx) * inv);
}

// ---------------- K5: out[b,n,h*96+e] = attn^T @ v + f_a (residual) ----------------
__global__ void out_gemm_kernel(const __bf16* __restrict__ attnT, const __bf16* __restrict__ vt,
                                const float* __restrict__ f_a, float* __restrict__ out) {
    int wid  = (blockIdx.x * blockDim.x + threadIdx.x) >> 5;
    int lane = threadIdx.x & 31;
    const int TILES = 16 * 6;
    if (wid >= BB * HH * TILES) return;
    int bh = wid / TILES, t = wid % TILES;
    int b = bh / HH, h = bh % HH;
    int nt = t / 6, et = t % 6;
    const __bf16* A  = attnT + (size_t)bh * KKE * KKE;  // [n][m]
    const __bf16* Bt = vt    + (size_t)bh * DKK * KKE;  // [e][m]
    v8f acc = {};
#pragma unroll
    for (int k0 = 0; k0 < KKE; k0 += 32) {
        v16bf a  = load_a_frag(A, KKE, nt * 16, k0, lane);
        v16bf bb = load_b_frag(Bt, KKE, et * 16, k0, lane);
        acc = __builtin_amdgcn_wmma_f32_16x16x32_bf16(false, a, false, bb, (short)0, acc, false, false);
    }
    int e     = et * 16 + (lane & 15);
    int nbase = nt * 16 + ((lane & 16) ? 8 : 0);
#pragma unroll
    for (int j = 0; j < 8; ++j) {
        size_t idx = ((size_t)b * KKE + nbase + j) * DD + h * DKK + e;
        out[idx] = acc[j] + f_a[idx];
    }
}

// ---------------- host launcher ----------------
extern "C" void kernel_launch(void* const* d_in, const int* in_sizes, int n_in,
                              void* d_out, int out_size, void* d_ws, size_t ws_size,
                              hipStream_t stream) {
    const float* f_a  = (const float*)d_in[0];
    const float* pe   = (const float*)d_in[1];
    const float* WG_w = (const float*)d_in[2];
    const float* WG_b = (const float*)d_in[3];
    const float* WK_w = (const float*)d_in[4];
    const float* WK_b = (const float*)d_in[5];
    const float* WQ_w = (const float*)d_in[6];
    const float* WQ_b = (const float*)d_in[7];
    const float* WV_w = (const float*)d_in[8];
    const float* WV_b = (const float*)d_in[9];
    float* out = (float*)d_out;
    char*  ws  = (char*)d_ws;

    // workspace layout (bytes), ~41.3 MB total
    constexpr size_t sz_fabf = (size_t)BB * KKE * DD * 2;
    constexpr size_t sz_wt   = (size_t)HH * DKK * DD * 2;
    constexpr size_t sz_kqv  = (size_t)BB * HH * KKE * DKK * 2;
    constexpr size_t sz_log  = (size_t)BB * HH * KKE * KKE * 4;
    constexpr size_t off_fabf = 0;
    constexpr size_t off_wkt  = off_fabf + sz_fabf;
    constexpr size_t off_wqt  = off_wkt + sz_wt;
    constexpr size_t off_wvt  = off_wqt + sz_wt;
    constexpr size_t off_k    = off_wvt + sz_wt;
    constexpr size_t off_q    = off_k + sz_kqv;
    constexpr size_t off_vt   = off_q + sz_kqv;
    constexpr size_t off_log  = off_vt + sz_kqv;
    constexpr size_t off_at   = off_log + sz_log;

    __bf16* fabf  = (__bf16*)(ws + off_fabf);
    __bf16* wkt   = (__bf16*)(ws + off_wkt);
    __bf16* wqt   = (__bf16*)(ws + off_wqt);
    __bf16* wvt   = (__bf16*)(ws + off_wvt);
    __bf16* kbf   = (__bf16*)(ws + off_k);
    __bf16* qbf   = (__bf16*)(ws + off_q);
    __bf16* vtbf  = (__bf16*)(ws + off_vt);
    float*  logit = (float*)(ws + off_log);
    __bf16* attnT = (__bf16*)(ws + off_at);

    // casts / transposes
    cast_bf16_kernel<<<(BB * KKE * DD + 255) / 256, 256, 0, stream>>>(f_a, fabf, BB * KKE * DD);
    wtrans_bf16_kernel<<<(HH * DD * DKK + 255) / 256, 256, 0, stream>>>(WK_w, wkt);
    wtrans_bf16_kernel<<<(HH * DD * DKK + 255) / 256, 256, 0, stream>>>(WQ_w, wqt);
    wtrans_bf16_kernel<<<(HH * DD * DKK + 255) / 256, 256, 0, stream>>>(WV_w, wvt);

    // geo bias (writes log-clipped relu into logits buffer)
    geo_bias_kernel<<<(BB * KKE * KKE) / 256, 256, 0, stream>>>(pe, WG_w, WG_b, logit);

    // q/k/v projections: 3*B*H*96 wave-tiles, 8 waves per 256-thread block
    {
        int waves = 3 * BB * HH * 96;
        qkv_gemm_kernel<<<(waves * 32 + 255) / 256, 256, 0, stream>>>(
            fabf, wkt, wqt, wvt, WK_b, WQ_b, WV_b, kbf, qbf, vtbf);
    }
    // scores accumulated onto geo bias
    {
        int waves = BB * HH * 256;
        scores_kernel<<<(waves * 32 + 255) / 256, 256, 0, stream>>>(kbf, qbf, logit);
    }
    // column softmax -> attn^T bf16
    softmax_kernel<<<BB * HH, 256, 0, stream>>>(logit, attnT);
    // output GEMM + residual
    {
        int waves = BB * HH * 96;
        out_gemm_kernel<<<(waves * 32 + 255) / 256, 256, 0, stream>>>(attnT, vtbf, f_a, out);
    }
}
